// Model_7773890806419
// MI455X (gfx1250) — matrix operand order repaired and checked by reading, not compiled, not run on hardware
//
#include <hip/hip_runtime.h>
#include <hip/hip_bf16.h>

typedef __attribute__((ext_vector_type(16))) _Float16 v16h;
typedef __attribute__((ext_vector_type(8)))  _Float16 v8h;
typedef __attribute__((ext_vector_type(8)))  float    v8f;

#define UNITS 1024   // B
#define HID   256    // U

__device__ __forceinline__ float sigm_f(float x) {
    return 1.0f / (1.0f + __expf(-x));
}
__device__ __forceinline__ float tanh_f(float x) {
    x = fminf(fmaxf(x, -15.0f), 15.0f);
    float e = __expf(2.0f * x);
    return (e - 1.0f) / (e + 1.0f);
}

// ---------------------------------------------------------------------------
// Prepack gate weights [K_orig, 256] f32 -> WMMA B-layout f16, K padded.
// out index = ((((utile*ktiles + ktile)*4 + g)*32 + lane)*16 + e)
// B 32x16 tile layout: lane L -> N = L&15, K = e<8 ? (L>>4)*8 + e : 16 + (L>>4)*8 + (e-8)
// ---------------------------------------------------------------------------
__global__ void prepack_w(const float* __restrict__ wf, const float* __restrict__ wi,
                          const float* __restrict__ wc, const float* __restrict__ wo,
                          int k_orig, int ktiles, _Float16* __restrict__ out) {
    int idx = blockIdx.x * blockDim.x + threadIdx.x;
    int total = 16 * ktiles * 4 * 32 * 16;
    if (idx >= total) return;
    int e     = idx & 15;
    int lane  = (idx >> 4) & 31;
    int g     = (idx >> 9) & 3;
    int rest  = idx >> 11;
    int ktile = rest % ktiles;
    int utile = rest / ktiles;
    int k = ktile * 32 + ((e < 8) ? (lane >> 4) * 8 + e : 16 + (lane >> 4) * 8 + (e - 8));
    int u = utile * 16 + (lane & 15);
    const float* w = (g == 0) ? wf : (g == 1) ? wi : (g == 2) ? wc : wo;
    float v = (k < k_orig) ? w[(size_t)k * HID + u] : 0.0f;
    out[idx] = (_Float16)v;
}

__global__ void pack_bias(const float* __restrict__ bf, const float* __restrict__ bi,
                          const float* __restrict__ bc, const float* __restrict__ bo,
                          float* __restrict__ out) {
    int i = blockIdx.x * blockDim.x + threadIdx.x;
    if (i >= 4 * HID) return;
    int g = i >> 8, u = i & 255;
    const float* b = (g == 0) ? bf : (g == 1) ? bi : (g == 2) ? bc : bo;
    out[i] = b[u];
}

// X[t][b][j] = emb[tokens[b][t]][j] (f16), j in [0,128), zero-padded past 100
__global__ void gather_x(const int* __restrict__ tokens, const float* __restrict__ emb,
                         _Float16* __restrict__ X) {
    size_t idx = (size_t)blockIdx.x * blockDim.x + threadIdx.x;
    if (idx >= (size_t)80 * UNITS * 128) return;
    int j = (int)(idx & 127);
    int b = (int)((idx >> 7) & 1023);
    int t = (int)(idx >> 17);
    float v = 0.0f;
    if (j < 100) {
        int tok = tokens[(size_t)b * 80 + t];
        v = emb[(size_t)tok * 100 + j];
    }
    X[idx] = (_Float16)v;
}

__global__ void zero_u32(unsigned* __restrict__ p, int n) {
    int i = blockIdx.x * blockDim.x + threadIdx.x;
    if (i < n) p[i] = 0u;
}

// ---------------------------------------------------------------------------
// Fused LSTM cell step. One wave (32 threads) computes TWO 16-row M-tiles
// for one 16-col U-tile, all four gates, sharing a single B-tile fetch.
// A (hx) = [Ah(256 cols) | Ax(XS cols)] f16 row-major.
// grid = (32 m-pairs, 16 u-tiles), block = 32.
// ---------------------------------------------------------------------------
template <int KX, int XS>
__global__ __launch_bounds__(32) void lstm_cell(
    const _Float16* __restrict__ Ah,    // [1024, 256] h_prev (f16)
    const _Float16* __restrict__ Ax,    // [1024, XS]  x_t or h1 (f16)
    const _Float16* __restrict__ Bp,    // prepacked gate weights
    const float*    __restrict__ bias,  // [4*256] packed
    float*          __restrict__ c,     // [1024, 256] cell state in/out
    _Float16*       __restrict__ hout)  // [1024, 256] new h (f16)
{
    const int lane = threadIdx.x;
    const int mt = blockIdx.x, ut = blockIdx.y;
    const int hl = lane >> 4;        // half of wave
    const int lr = lane & 15;
    const int m_base = mt * 32;      // two 16-row tiles
    const int kt_total = 8 + KX;

    v8f acc[2][4];
    #pragma unroll
    for (int mi = 0; mi < 2; ++mi)
        #pragma unroll
        for (int g = 0; g < 4; ++g)
            acc[mi][g] = (v8f){};

    union AV { v16h v; v8h h[2]; };

    // K tiles 0..7 from h-part
    #pragma unroll
    for (int kt = 0; kt < 8; ++kt) {
        const _Float16* bpk = Bp + ((size_t)(ut * kt_total + kt) * 4) * 512 + lane * 16;
        v16h b0 = *(const v16h*)(bpk);
        v16h b1 = *(const v16h*)(bpk + 512);
        v16h b2 = *(const v16h*)(bpk + 1024);
        v16h b3 = *(const v16h*)(bpk + 1536);
        AV a[2];
        #pragma unroll
        for (int mi = 0; mi < 2; ++mi) {
            const _Float16* ap = Ah + (size_t)(m_base + mi * 16 + lr) * 256 + kt * 32;
            a[mi].h[0] = *(const v8h*)(ap + hl * 8);       // elems 0..7  : K = 8*hl+0..7
            a[mi].h[1] = *(const v8h*)(ap + 16 + hl * 8);  // elems 8..15 : K = 16+8*hl+0..7
        }
        #pragma unroll
        for (int mi = 0; mi < 2; ++mi) {
            acc[mi][0] = __builtin_amdgcn_wmma_f32_16x16x32_f16(false, a[mi].v, false, b0, (short)0, acc[mi][0], false, false);
            acc[mi][1] = __builtin_amdgcn_wmma_f32_16x16x32_f16(false, a[mi].v, false, b1, (short)0, acc[mi][1], false, false);
            acc[mi][2] = __builtin_amdgcn_wmma_f32_16x16x32_f16(false, a[mi].v, false, b2, (short)0, acc[mi][2], false, false);
            acc[mi][3] = __builtin_amdgcn_wmma_f32_16x16x32_f16(false, a[mi].v, false, b3, (short)0, acc[mi][3], false, false);
        }
    }
    // K tiles from x-part
    #pragma unroll
    for (int kt = 0; kt < KX; ++kt) {
        const _Float16* bpk = Bp + ((size_t)(ut * kt_total + 8 + kt) * 4) * 512 + lane * 16;
        v16h b0 = *(const v16h*)(bpk);
        v16h b1 = *(const v16h*)(bpk + 512);
        v16h b2 = *(const v16h*)(bpk + 1024);
        v16h b3 = *(const v16h*)(bpk + 1536);
        AV a[2];
        #pragma unroll
        for (int mi = 0; mi < 2; ++mi) {
            const _Float16* ap = Ax + (size_t)(m_base + mi * 16 + lr) * XS + kt * 32;
            a[mi].h[0] = *(const v8h*)(ap + hl * 8);
            a[mi].h[1] = *(const v8h*)(ap + 16 + hl * 8);
        }
        #pragma unroll
        for (int mi = 0; mi < 2; ++mi) {
            acc[mi][0] = __builtin_amdgcn_wmma_f32_16x16x32_f16(false, a[mi].v, false, b0, (short)0, acc[mi][0], false, false);
            acc[mi][1] = __builtin_amdgcn_wmma_f32_16x16x32_f16(false, a[mi].v, false, b1, (short)0, acc[mi][1], false, false);
            acc[mi][2] = __builtin_amdgcn_wmma_f32_16x16x32_f16(false, a[mi].v, false, b2, (short)0, acc[mi][2], false, false);
            acc[mi][3] = __builtin_amdgcn_wmma_f32_16x16x32_f16(false, a[mi].v, false, b3, (short)0, acc[mi][3], false, false);
        }
    }

    // Elementwise LSTM update on D layout: lane -> N = lr, vgpr v -> M = 8*hl + v
    const int n = ut * 16 + lr;
    const float bF = bias[n], bI = bias[256 + n], bC = bias[512 + n], bO = bias[768 + n];
    #pragma unroll
    for (int mi = 0; mi < 2; ++mi) {
        #pragma unroll
        for (int v = 0; v < 8; ++v) {
            int m = m_base + mi * 16 + hl * 8 + v;
            float gf = sigm_f(acc[mi][0][v] + bF);
            float gi = sigm_f(acc[mi][1][v] + bI);
            float ct = tanh_f(acc[mi][2][v] + bC);
            float go = sigm_f(acc[mi][3][v] + bO);
            size_t off = (size_t)m * HID + n;
            float cn = gf * c[off] + gi * ct;
            c[off] = cn;
            hout[off] = (_Float16)(go * tanh_f(cn));
        }
    }
}

// out[b] = sigmoid(h2[b,:] . w_out + b_out)
__global__ void out_head(const _Float16* __restrict__ h2, const float* __restrict__ w,
                         const float* __restrict__ b, float* __restrict__ out) {
    int i = blockIdx.x * blockDim.x + threadIdx.x;
    if (i >= UNITS) return;
    float s = b[0];
    #pragma unroll 4
    for (int j = 0; j < HID; ++j) s += (float)h2[(size_t)i * HID + j] * w[j];
    out[i] = sigm_f(s);
}

extern "C" void kernel_launch(void* const* d_in, const int* in_sizes, int n_in,
                              void* d_out, int out_size, void* d_ws, size_t ws_size,
                              hipStream_t stream) {
    const int*   tokens = (const int*)  d_in[0];
    const float* emb    = (const float*)d_in[1];
    const float* wf1 = (const float*)d_in[2];  const float* bf1 = (const float*)d_in[3];
    const float* wi1 = (const float*)d_in[4];  const float* bi1 = (const float*)d_in[5];
    const float* wc1 = (const float*)d_in[6];  const float* bc1 = (const float*)d_in[7];
    const float* wo1 = (const float*)d_in[8];  const float* bo1 = (const float*)d_in[9];
    const float* wf2 = (const float*)d_in[10]; const float* bf2 = (const float*)d_in[11];
    const float* wi2 = (const float*)d_in[12]; const float* bi2 = (const float*)d_in[13];
    const float* wc2 = (const float*)d_in[14]; const float* bc2 = (const float*)d_in[15];
    const float* wo2 = (const float*)d_in[16]; const float* bo2 = (const float*)d_in[17];
    const float* w_out = (const float*)d_in[18];
    const float* b_out = (const float*)d_in[19];

    // ---- carve workspace (256B-aligned chunks) ----
    char* p = (char*)d_ws;
    auto carve = [&](size_t bytes) -> void* {
        void* r = (void*)p;
        p += (bytes + 255) & ~(size_t)255;
        return r;
    };
    _Float16* X    = (_Float16*)carve((size_t)80 * UNITS * 128 * 2);  // 20 MB
    _Float16* H1a  = (_Float16*)carve((size_t)UNITS * HID * 2);
    _Float16* H1b  = (_Float16*)carve((size_t)UNITS * HID * 2);
    _Float16* H2a  = (_Float16*)carve((size_t)UNITS * HID * 2);
    _Float16* H2b  = (_Float16*)carve((size_t)UNITS * HID * 2);
    float*    c1   = (float*)   carve((size_t)UNITS * HID * 4);
    float*    c2   = (float*)   carve((size_t)UNITS * HID * 4);
    _Float16* Bp1  = (_Float16*)carve((size_t)16 * 12 * 4 * 512 * 2);
    _Float16* Bp2  = (_Float16*)carve((size_t)16 * 16 * 4 * 512 * 2);
    float*    bia1 = (float*)   carve(4 * HID * 4);
    float*    bia2 = (float*)   carve(4 * HID * 4);

    // ---- one-time prep (re-run every call; deterministic) ----
    {
        int tot1 = 16 * 12 * 4 * 32 * 16;
        prepack_w<<<(tot1 + 255) / 256, 256, 0, stream>>>(wf1, wi1, wc1, wo1, 356, 12, Bp1);
        int tot2 = 16 * 16 * 4 * 32 * 16;
        prepack_w<<<(tot2 + 255) / 256, 256, 0, stream>>>(wf2, wi2, wc2, wo2, 512, 16, Bp2);
        pack_bias<<<4, 256, 0, stream>>>(bf1, bi1, bc1, bo1, bia1);
        pack_bias<<<4, 256, 0, stream>>>(bf2, bi2, bc2, bo2, bia2);
        size_t totx = (size_t)80 * UNITS * 128;
        gather_x<<<(unsigned)((totx + 255) / 256), 256, 0, stream>>>(tokens, emb, X);
        int nh = UNITS * HID * 2 / 4;   // f16 buffer in u32 words
        int nc = UNITS * HID;           // f32 buffer in u32 words
        zero_u32<<<(nh + 255) / 256, 256, 0, stream>>>((unsigned*)H1a, nh);
        zero_u32<<<(nh + 255) / 256, 256, 0, stream>>>((unsigned*)H2a, nh);
        zero_u32<<<(nc + 255) / 256, 256, 0, stream>>>((unsigned*)c1, nc);
        zero_u32<<<(nc + 255) / 256, 256, 0, stream>>>((unsigned*)c2, nc);
    }

    // ---- recurrent loop: stream order provides the inter-step sync ----
    _Float16 *h1s = H1a, *h1d = H1b, *h2s = H2a, *h2d = H2b;
    dim3 grid(UNITS / 32, HID / 16);   // (32 m-pairs, 16 u-tiles)
    dim3 blk(32);
    for (int t = 0; t < 80; ++t) {
        lstm_cell<4, 128><<<grid, blk, 0, stream>>>(
            h1s, X + (size_t)t * UNITS * 128, Bp1, bia1, c1, h1d);
        lstm_cell<8, 256><<<grid, blk, 0, stream>>>(
            h2s, h1d, Bp2, bia2, c2, h2d);
        _Float16* tmp;
        tmp = h1s; h1s = h1d; h1d = tmp;
        tmp = h2s; h2s = h2d; h2d = tmp;
    }

    out_head<<<4, 256, 0, stream>>>(h2s, w_out, b_out, (float*)d_out);
}
